// EF_42984032699109
// MI455X (gfx1250) — matrix operand order repaired and checked by reading, not compiled, not run on hardware
//
#include <hip/hip_runtime.h>
#include <math.h>

// ---------------------------------------------------------------------------
// Fixed problem sizes from the reference
// ---------------------------------------------------------------------------
#define FDIM   32
#define RDIM   16
#define NBAS   16
#define AATOM  60
#define NBATCH 32
#define NATOM  (NBATCH*AATOM)          // 1920
#define EDGE_PER (AATOM-1)             // 59
#define NEDGE  (NATOM*EDGE_PER)        // 113280
#define XTILE  (RDIM*FDIM)             // 512 floats per atom
#define XSZ    (NATOM*XTILE)           // 983040
#define KEC    7.199822675975274f
#define CUTR   6.0f
#define OUTSZ  (NBATCH + NATOM*3 + NATOM + NBATCH) // 7744

typedef __attribute__((ext_vector_type(16))) _Float16 v16h;
typedef __attribute__((ext_vector_type(8)))  float    v8f;

// ---------------------------------------------------------------------------
// CDNA5 async global->LDS staging (ASYNCcnt path), with portable fallback.
// ---------------------------------------------------------------------------
#if defined(__AMDGCN__) && __has_builtin(__builtin_amdgcn_global_load_async_to_lds_b32) && __has_builtin(__builtin_amdgcn_s_wait_asynccnt)
#define HAVE_ASYNC 1
#else
#define HAVE_ASYNC 0
#endif

__device__ __forceinline__ void cp1_async(float* d, const float* s) {
#if HAVE_ASYNC
  __builtin_amdgcn_global_load_async_to_lds_b32(
      (__attribute__((address_space(1))) int*)(s),
      (__attribute__((address_space(3))) int*)(d), 0, 0);
#else
  *d = *s;
#endif
}
__device__ __forceinline__ void cp_wait_async() {
#if HAVE_ASYNC
  __builtin_amdgcn_s_wait_asynccnt(0);
#endif
}

__device__ __forceinline__ int degf(int r) { return (r == 0) ? 0 : (r < 4) ? 1 : (r < 9) ? 2 : 3; }

__device__ __forceinline__ float wsum32(float v) {
  #pragma unroll
  for (int o = 16; o > 0; o >>= 1) v += __shfl_xor(v, o, 32);
  return v;
}

// ---------------------------------------------------------------------------
// Spherical harmonics (16 comps, L<=3) and Jacobian contraction
// ---------------------------------------------------------------------------
__device__ __forceinline__ void sph16(float x, float y, float z, float* Y) {
  float x2 = x * x, y2 = y * y, z2 = z * z;
  Y[0] = 1.f;  Y[1] = y;      Y[2] = z;          Y[3] = x;
  Y[4] = x * y; Y[5] = y * z; Y[6] = 3.f * z2 - 1.f; Y[7] = x * z;
  Y[8] = x2 - y2; Y[9] = y * (3.f * x2 - y2); Y[10] = x * y * z;
  Y[11] = y * (5.f * z2 - 1.f); Y[12] = z * (5.f * z2 - 3.f);
  Y[13] = x * (5.f * z2 - 1.f); Y[14] = z * (x2 - y2); Y[15] = x * (x2 - 3.f * y2);
}

__device__ __forceinline__ void sph16_jac(float x, float y, float z, const float* g,
                                          float* gx, float* gy, float* gz) {
  float x2 = x * x, y2 = y * y, z2 = z * z;
  *gx = g[3] + g[4] * y + g[7] * z + g[8] * 2.f * x + g[9] * 6.f * x * y + g[10] * y * z
      + g[13] * (5.f * z2 - 1.f) + g[14] * 2.f * x * z + g[15] * (3.f * x2 - 3.f * y2);
  *gy = g[1] + g[4] * x + g[5] * z - g[8] * 2.f * y + g[9] * (3.f * x2 - 3.f * y2)
      + g[10] * x * z + g[11] * (5.f * z2 - 1.f) - g[14] * 2.f * y * z - g[15] * 6.f * x * y;
  *gz = g[2] + g[5] * y + g[6] * 6.f * z + g[7] * x + g[10] * x * y + g[11] * 10.f * y * z
      + g[12] * (15.f * z2 - 3.f) + g[13] * 10.f * x * z + g[14] * (x2 - y2);
}

// ---------------------------------------------------------------------------
// Bernstein radial basis (NB=16) with envelope; value and d/dr versions
// ---------------------------------------------------------------------------
__device__ __forceinline__ void radial16(float r, float* rad) {
  const float C[16] = {1.f,15.f,105.f,455.f,1365.f,3003.f,5005.f,6435.f,
                       6435.f,5005.f,3003.f,1365.f,455.f,105.f,15.f,1.f};
  float u = 1.f / (1.f + r), om = 1.f - u;
  float t = (r * r) / (CUTR * CUTR);
  float cut = (r < CUTR) ? __expf(-t / fmaxf(1.f - t, 1e-9f)) : 0.f;
  float up[16], op[16];
  up[0] = 1.f; op[0] = 1.f;
  #pragma unroll
  for (int k = 1; k < 16; ++k) { up[k] = up[k - 1] * u; op[k] = op[k - 1] * om; }
  #pragma unroll
  for (int k = 0; k < 16; ++k) rad[k] = C[k] * up[k] * op[15 - k] * cut;
}

__device__ __forceinline__ void radial16_d(float r, float* rad, float* drad) {
  const float C[16] = {1.f,15.f,105.f,455.f,1365.f,3003.f,5005.f,6435.f,
                       6435.f,5005.f,3003.f,1365.f,455.f,105.f,15.f,1.f};
  float u = 1.f / (1.f + r), om = 1.f - u;
  float t = (r * r) / (CUTR * CUTR);
  float cut = 0.f, dcut = 0.f;
  if (r < CUTR) {
    float m = fmaxf(1.f - t, 1e-9f);
    cut = __expf(-t / m);
    float dexp_dt = (1.f - t > 1e-9f) ? (-1.f / (m * m)) : (-1.f / m);
    dcut = cut * dexp_dt * (2.f * r / (CUTR * CUTR));
  }
  float up[16], op[16];
  up[0] = 1.f; op[0] = 1.f;
  #pragma unroll
  for (int k = 1; k < 16; ++k) { up[k] = up[k - 1] * u; op[k] = op[k - 1] * om; }
  float dudr = -u * u;
  #pragma unroll
  for (int k = 0; k < 16; ++k) {
    float b = C[k] * up[k] * op[15 - k];
    float db = C[k] * ((k > 0 ? k * up[k - 1] * op[15 - k] : 0.f)
                      - (k < 15 ? (15 - k) * up[k] * op[14 - k] : 0.f));
    rad[k] = b * cut;
    drad[k] = db * dudr * cut + b * dcut;
  }
}

// switch(x; 0, 10) value + derivative wrt x
__device__ __forceinline__ float switch_fn(float x, float* dsw_dx) {
  float t = x * 0.1f, dt = 0.1f;
  if (t < 1e-9f)       { t = 1e-9f;       dt = 0.f; }
  if (t > 1.f - 1e-9f) { t = 1.f - 1e-9f; dt = 0.f; }
  float f = __expf(-1.f / t);
  float g = __expf(-1.f / (1.f - t));
  float sw = f / (f + g);
  if (dsw_dx) *dsw_dx = sw * (1.f - sw) * (1.f / (t * t) + 1.f / ((1.f - t) * (1.f - t))) * dt;
  return sw;
}

// ---------------------------------------------------------------------------
// WMMA fragment helpers (wave32, 16x16x32 f16 -> f32).
// A: 16x32 f32 tile (row-major, stride 32) with optional per-column scale.
// B: 32x32 f32 weight, selected 32x16 column panel (n0 in {0,16}); optional ^T.
// ---------------------------------------------------------------------------
__device__ __forceinline__ v16h frag_A(const float* tile, int lane, const float* colscale) {
  int row = lane & 15;
  int kofs = ((lane >> 4) & 1) * 8;
  v16h a;
  #pragma unroll
  for (int j = 0; j < 16; ++j) {
    int k = (j < 8 ? j : j + 8) + kofs;
    float v = tile[row * 32 + k];
    if (colscale) v *= colscale[k];
    a[j] = (_Float16)v;
  }
  return a;
}

__device__ __forceinline__ v16h frag_B(const float* W, int lane, int n0, int transposeW) {
  int col = n0 + (lane & 15);
  int kbase = ((lane >> 4) & 1) * 8;
  v16h b;
  #pragma unroll
  for (int j = 0; j < 16; ++j) {
    int k = (j < 8 ? j : j + 8) + kbase;
    float v = transposeW ? W[col * 32 + k] : W[k * 32 + col];
    b[j] = (_Float16)v;
  }
  return b;
}

__device__ __forceinline__ void store_D(float* tile, int lane, int n0, v8f d, const float* bias) {
  int col = n0 + (lane & 15);
  int rbase = ((lane >> 4) & 1) * 8;
  float bv = bias ? bias[col] : 0.f;          // hoisted; branchless row-0 add below
  #pragma unroll
  for (int v = 0; v < 8; ++v) {
    int m = rbase + v;
    float val = d[v] + ((m == 0) ? bv : 0.f);
    tile[m * 32 + col] = val;
  }
}

// 16x32 tile  x  32x32 weight  ->  16x32 tile (two v_wmma_f32_16x16x32_f16)
__device__ __forceinline__ void dense16(const float* in, float* out, const float* W,
                                        const float* bias, const float* gate,
                                        int lane, int transposeW) {
  v16h a  = frag_A(in, lane, gate);
  v16h b0 = frag_B(W, lane, 0, transposeW);
  v16h b1 = frag_B(W, lane, 16, transposeW);
  v8f c0 = {}; v8f c1 = {};
  c0 = __builtin_amdgcn_wmma_f32_16x16x32_f16(false, a, false, b0, (short)0, c0, false, false);
  c1 = __builtin_amdgcn_wmma_f32_16x16x32_f16(false, a, false, b1, (short)0, c1, false, false);
  store_D(out, lane, 0, c0, bias);
  store_D(out, lane, 16, c1, bias);
}

// ---------------------------------------------------------------------------
// Utility kernels
// ---------------------------------------------------------------------------
__global__ void k_fill0(float* p, int n) {
  int i = blockIdx.x * blockDim.x + threadIdx.x;
  if (i < n) p[i] = 0.f;
}

// Edge geometry: disp, r, Y16, rad16, d(=|disp+off|), Rij
__global__ void k_edges(const float* __restrict__ pos, const float* __restrict__ bm,
                        const int* __restrict__ dstv, const int* __restrict__ srcv,
                        float* __restrict__ dispb, float* __restrict__ rb,
                        float* __restrict__ Yb, float* __restrict__ radb,
                        float* __restrict__ db, float* __restrict__ Rb) {
  int e = blockIdx.x * blockDim.x + threadIdx.x;
  if (e >= NEDGE) return;
  int s = srcv[e], d0 = dstv[e];
  float dx = pos[s * 3 + 0] - pos[d0 * 3 + 0];
  float dy = pos[s * 3 + 1] - pos[d0 * 3 + 1];
  float dz = pos[s * 3 + 2] - pos[d0 * 3 + 2];
  float r = sqrtf(dx * dx + dy * dy + dz * dz);
  float ir = 1.f / fmaxf(r, 1e-9f);
  float Y[16]; sph16(dx * ir, dy * ir, dz * ir, Y);
  float rad[16]; radial16(r, rad);
  #pragma unroll
  for (int k = 0; k < 16; ++k) { Yb[e * 16 + k] = Y[k]; radb[e * 16 + k] = rad[k]; }
  dispb[e * 3 + 0] = dx; dispb[e * 3 + 1] = dy; dispb[e * 3 + 2] = dz;
  rb[e] = r;
  float off = 1.f - bm[e];
  float ax = dx + off, ay = dy + off, az = dz + off;
  float dd = sqrtf(ax * ax + ay * ay + az * az);
  float sw = switch_fn(2.f * dd, nullptr);
  float Rij = sw * rsqrtf(dd * dd + 1.f) + (1.f - sw) / fmaxf(dd, 1e-9f);
  db[e] = dd; Rb[e] = Rij;
}

// x_emb[n][r][f] = (r==0) ? embed[z[n]][f] : 0
__global__ void k_init_x(const float* __restrict__ embed, const int* __restrict__ zi,
                         float* __restrict__ x_emb) {
  int i = blockIdx.x * blockDim.x + threadIdx.x;
  if (i >= XSZ) return;
  int n = i >> 9, rem = i & 511, r = rem >> 5, f = rem & 31;
  x_emb[i] = (r == 0) ? embed[zi[n] * FDIM + f] : 0.f;
}

// ---------------------------------------------------------------------------
// Fused message passing (one dst atom per block; 59 contiguous edges).
// mp weights async-staged to LDS once; rad1/rad2 built cooperatively per edge.
// ---------------------------------------------------------------------------
__global__ __launch_bounds__(256) void k_msg(const float* __restrict__ xprev, const float* __restrict__ Yb,
                      const float* __restrict__ radb, const int* __restrict__ srcv,
                      const float* __restrict__ w1, const float* __restrict__ w2,
                      float* __restrict__ xout, int last) {
  __shared__ float r12[256];   // [which(2)][l(4)][f(32)]
  __shared__ float red[512];
  __shared__ float wsh[2048];  // mp_w1 (1024) + mp_w2 (1024) for this iteration
  int n = blockIdx.x, t = threadIdx.x;
  for (int i = t; i < 1024; i += 256) { cp1_async(&wsh[i], &w1[i]); cp1_async(&wsh[1024 + i], &w2[i]); }
  cp_wait_async();
  __syncthreads();
  int r0 = t >> 5, f = t & 31;
  float acc0 = 0.f, acc1 = 0.f;
  for (int e0 = 0; e0 < EDGE_PER; ++e0) {
    int e = n * EDGE_PER + e0;
    __syncthreads();
    {
      int which = t >> 7, l = (t >> 5) & 3, ff = t & 31;
      const float* rp = radb + e * 16;
      float s = 0.f;
      #pragma unroll
      for (int b = 0; b < 16; ++b) s += rp[b] * wsh[which * 1024 + (b * 4 + l) * 32 + ff];
      r12[t] = s;
    }
    if (e0 + 1 < EDGE_PER)  // prefetch next source tile (global_prefetch_b8)
      __builtin_prefetch(xprev + (size_t)srcv[e + 1] * XTILE + t, 0, 3);
    __syncthreads();
    const float* xs = xprev + (size_t)srcv[e] * XTILE;
    const float* Ye = Yb + e * 16;
    float xs0 = xs[f];
    {
      int r = r0; int l = degf(r);
      float yv = Ye[r], r1 = r12[l * 32 + f];
      float xsr = xs[r * 32 + f];
      if (!last) acc0 += yv * r1 * xs0 + xsr * r12[128 + l * 32 + f];
      else       acc0 += xsr * yv * r1 + ((r == 0) ? xs0 * r12[128 + f] : 0.f);
    }
    {
      int r = r0 + 8; int l = degf(r);
      float yv = Ye[r], r1 = r12[l * 32 + f];
      float xsr = xs[r * 32 + f];
      if (!last) acc1 += yv * r1 * xs0 + xsr * r12[128 + l * 32 + f];
      else       acc1 += xsr * yv * r1;
    }
  }
  float* xo = xout + (size_t)n * XTILE;
  if (!last) {
    xo[t] = acc0; xo[t + 256] = acc1;
  } else {
    red[t] = acc0; red[t + 256] = acc1;
    __syncthreads();
    float v0 = 0.f;
    if (r0 == 0) {
      #pragma unroll
      for (int r = 0; r < 16; ++r) v0 += red[r * 32 + f];
    }
    xo[t] = (r0 == 0) ? v0 : 0.f;
    xo[t + 256] = 0.f;
  }
}

// ---------------------------------------------------------------------------
// Per-atom MLP forward with WMMA (4 waves/block, one atom per wave).
// Iteration weights async-staged to LDS once per block (~29 KB).
// ---------------------------------------------------------------------------
#define MLPW 4
#define WSH_RESW 0         // 6*1024
#define WSH_RESB 6144      // 6*32
#define WSH_PW   6336      // 1024
#define WSH_PB   7360      // 32
#define WSH_SZ   7392
__global__ __launch_bounds__(128) void k_mlp(const float* __restrict__ xin, float* __restrict__ xout,
                      const float* __restrict__ resw, const float* __restrict__ resb,
                      const float* __restrict__ postw, const float* __restrict__ postb, int iter) {
  __shared__ float xt[MLPW][512], yt[MLPW][512], wt[MLPW][512], gt[MLPW][32];
  __shared__ float wsh[WSH_SZ];
  int tid = threadIdx.x;
  {
    const float* rw = resw + (size_t)iter * 6144;
    const float* rbp = resb + (size_t)iter * 192;
    const float* pw = postw + (size_t)iter * 1024;
    const float* pb = postb + (size_t)iter * 32;
    for (int i = tid; i < 6144; i += 128) cp1_async(&wsh[WSH_RESW + i], &rw[i]);
    for (int i = tid; i < 192; i += 128)  cp1_async(&wsh[WSH_RESB + i], &rbp[i]);
    for (int i = tid; i < 1024; i += 128) cp1_async(&wsh[WSH_PW + i], &pw[i]);
    if (tid < 32) cp1_async(&wsh[WSH_PB + tid], &pb[tid]);
    cp_wait_async();
  }
  __syncthreads();
  int wv = tid >> 5, lane = tid & 31;
  int n = blockIdx.x * MLPW + wv;
  float* X = xt[wv]; float* Yt = yt[wv]; float* Wt = wt[wv]; float* G = gt[wv];
  const float* gx = xin + (size_t)n * XTILE;
  #pragma unroll
  for (int k = 0; k < 16; ++k) X[k * 32 + lane] = gx[k * 32 + lane];
  __syncthreads();
  for (int j = 0; j < 3; ++j) {
    const float* W0 = wsh + WSH_RESW + j * 2048;
    const float* B0 = wsh + WSH_RESB + j * 64;
    const float* W1 = W0 + 1024;
    const float* B1 = B0 + 32;
    G[lane] = 1.f / (1.f + __expf(-X[lane]));            // gsilu gate
    __syncthreads();
    dense16(X, Yt, W0, B0, G, lane, 0);
    __syncthreads();
    G[lane] = (Yt[lane] > 0.f) ? 1.f : 0.f;              // grelu gate
    __syncthreads();
    dense16(Yt, Wt, W1, B1, G, lane, 0);
    __syncthreads();
    #pragma unroll
    for (int k = 0; k < 16; ++k) X[k * 32 + lane] += Wt[k * 32 + lane];
    __syncthreads();
  }
  dense16(Wt, Yt, wsh + WSH_PW, wsh + WSH_PB, nullptr, lane, 0);
  __syncthreads();
  G[lane] = 1.f / (1.f + __expf(-Yt[lane]));
  __syncthreads();
  float* go = xout + (size_t)n * XTILE;
  #pragma unroll
  for (int k = 0; k < 16; ++k) go[k * 32 + lane] = X[k * 32 + lane] + Yt[k * 32 + lane] * G[lane];
}

// ---------------------------------------------------------------------------
// Readout: q, atomic energy ae, saved silu pre-activation
// ---------------------------------------------------------------------------
__global__ void k_readout(const float* __restrict__ x, const float* __restrict__ qk,
                          const float* __restrict__ cb, const float* __restrict__ ew1,
                          const float* __restrict__ ew2, const float* __restrict__ eb,
                          const int* __restrict__ zi, const float* __restrict__ am,
                          float* __restrict__ qb, float* __restrict__ ae, float* __restrict__ sbuf) {
  int n = blockIdx.x * blockDim.x + threadIdx.x;
  if (n >= NATOM) return;
  const float* x0 = x + (size_t)n * XTILE;
  float qr = 0.f, s = 0.f;
  #pragma unroll
  for (int f = 0; f < 32; ++f) { qr += x0[f] * qk[f]; s += x0[f] * ew1[f]; }
  int z = zi[n];
  qb[n] = (qr + cb[z]) * am[n];
  float sg = 1.f / (1.f + __expf(-s));
  ae[n] = (s * sg * ew2[0] + eb[z]) * am[n];
  sbuf[n] = s;
}

// Electrostatics: aele per atom, dE/dq accumulation, energy & bele outputs
__global__ void k_elec(const float* __restrict__ qb, const float* __restrict__ Rb,
                       const float* __restrict__ bm, const float* __restrict__ am,
                       const float* __restrict__ ae, const int* __restrict__ srcv,
                       const int* __restrict__ seg,
                       float* __restrict__ aele, float* __restrict__ gq,
                       float* __restrict__ out_energy, float* __restrict__ out_bele) {
  int n = blockIdx.x * blockDim.x + threadIdx.x;
  if (n >= NATOM) return;
  float qn = qb[n], amn = am[n], s1 = 0.f;
  for (int e0 = 0; e0 < EDGE_PER; ++e0) {
    int e = n * EDGE_PER + e0;
    int sa = srcv[e];
    float w = Rb[e] * bm[e];
    s1 += qb[sa] * w;
    atomicAdd(&gq[sa], KEC * amn * qn * w);
  }
  float an = KEC * qn * s1 * amn;
  atomicAdd(&gq[n], KEC * amn * s1);
  aele[n] = an;
  int b = seg[n];
  atomicAdd(&out_energy[b], ae[n] + an);
  atomicAdd(&out_bele[b], an);
}

// Head backward: grad wrt final x (row 0 only)
__global__ void k_head_bwd(const float* __restrict__ sbuf, const float* __restrict__ gq,
                           const float* __restrict__ qk, const float* __restrict__ ew1,
                           const float* __restrict__ ew2, const float* __restrict__ am,
                           float* __restrict__ g_a1) {
  int i = blockIdx.x * blockDim.x + threadIdx.x;
  if (i >= XSZ) return;
  int n = i >> 9, rem = i & 511, r = rem >> 5, f = rem & 31;
  float g = 0.f;
  if (r == 0) {
    float s = sbuf[n];
    float sg = 1.f / (1.f + __expf(-s));
    float dsilu = sg * (1.f + s * (1.f - sg));
    g = am[n] * (gq[n] * qk[f] + ew2[0] * dsilu * ew1[f]);
  }
  g_a1[i] = g;
}

// Electrostatic direct position gradient (through Rij(d))
__global__ void k_elec_bwd(const float* __restrict__ qb, const float* __restrict__ db,
                           const float* __restrict__ dispb, const float* __restrict__ bm,
                           const float* __restrict__ am, const int* __restrict__ srcv,
                           const int* __restrict__ dstv, float* __restrict__ gpos) {
  int e = blockIdx.x * blockDim.x + threadIdx.x;
  if (e >= NEDGE) return;
  float me = bm[e];
  int sa = srcv[e], da = dstv[e];
  float gfac = KEC * am[da] * qb[da] * qb[sa] * me;
  float off = 1.f - me;
  float ax = dispb[e * 3 + 0] + off, ay = dispb[e * 3 + 1] + off, az = dispb[e * 3 + 2] + off;
  float dd = db[e];
  float dsw; float sw = switch_fn(2.f * dd, &dsw); dsw *= 2.f;
  float p = rsqrtf(dd * dd + 1.f);
  float h = 1.f / fmaxf(dd, 1e-9f);
  float dp = -dd * p * p * p;
  float dh = (dd > 1e-9f) ? (-1.f / (dd * dd)) : 0.f;
  float gd = gfac * (dsw * (p - h) + sw * dp + (1.f - sw) * dh);
  float inv = 1.f / fmaxf(dd, 1e-9f);
  float gx = gd * ax * inv, gy = gd * ay * inv, gz = gd * az * inv;
  atomicAdd(&gpos[sa * 3 + 0], gx);  atomicAdd(&gpos[sa * 3 + 1], gy);  atomicAdd(&gpos[sa * 3 + 2], gz);
  atomicAdd(&gpos[da * 3 + 0], -gx); atomicAdd(&gpos[da * 3 + 1], -gy); atomicAdd(&gpos[da * 3 + 2], -gz);
}

// ---------------------------------------------------------------------------
// MLP backward: recompute forward in LDS, then reverse with transposed WMMA
// (2 waves/block, one atom/wave; weights async-staged to LDS)
// ---------------------------------------------------------------------------
#define MLPBW 2
__global__ __launch_bounds__(64) void k_mlp_bwd(const float* __restrict__ xin, const float* __restrict__ gin,
                          float* __restrict__ gout,
                          const float* __restrict__ resw, const float* __restrict__ resb,
                          const float* __restrict__ postw, const float* __restrict__ postb, int iter) {
  __shared__ float xs0[MLPBW][512], xs1[MLPBW][512], xs2[MLPBW][512];
  __shared__ float tts[MLPBW][512], gts[MLPBW][512], gas[MLPBW][512], wks[MLPBW][512];
  __shared__ float vrow[MLPBW][3][32], gateb[MLPBW][32], csb[MLPBW][32];
  __shared__ float wsh[WSH_SZ];
  int tid = threadIdx.x;
  {
    const float* rw = resw + (size_t)iter * 6144;
    const float* rbp = resb + (size_t)iter * 192;
    const float* pw = postw + (size_t)iter * 1024;
    const float* pb = postb + (size_t)iter * 32;
    for (int i = tid; i < 6144; i += 64) cp1_async(&wsh[WSH_RESW + i], &rw[i]);
    for (int i = tid; i < 192; i += 64)  cp1_async(&wsh[WSH_RESB + i], &rbp[i]);
    for (int i = tid; i < 1024; i += 64) cp1_async(&wsh[WSH_PW + i], &pw[i]);
    if (tid < 32) cp1_async(&wsh[WSH_PB + tid], &pb[tid]);
    cp_wait_async();
  }
  __syncthreads();
  int wv = tid >> 5, lane = tid & 31;
  int n = blockIdx.x * MLPBW + wv;
  float* XS[3] = { xs0[wv], xs1[wv], xs2[wv] };
  float* TT = tts[wv]; float* GT = gts[wv]; float* GA = gas[wv]; float* WK = wks[wv];
  float* G = gateb[wv]; float* CS = csb[wv];

  // ---- forward recompute, saving block inputs, grelu row0, post pre-act ----
  const float* gx = xin + (size_t)n * XTILE;
  #pragma unroll
  for (int k = 0; k < 16; ++k) XS[0][k * 32 + lane] = gx[k * 32 + lane];
  __syncthreads();
  #pragma unroll
  for (int j = 0; j < 3; ++j) {
    const float* W0 = wsh + WSH_RESW + j * 2048;
    const float* B0 = wsh + WSH_RESB + j * 64;
    const float* W1 = W0 + 1024;
    const float* B1 = B0 + 32;
    G[lane] = 1.f / (1.f + __expf(-XS[j][lane]));
    __syncthreads();
    dense16(XS[j], WK, W0, B0, G, lane, 0);
    __syncthreads();
    vrow[wv][j][lane] = WK[lane];
    G[lane] = (WK[lane] > 0.f) ? 1.f : 0.f;
    __syncthreads();
    dense16(WK, GA, W1, B1, G, lane, 0);           // GA = y_j
    __syncthreads();
    if (j < 2) {
      #pragma unroll
      for (int k = 0; k < 16; ++k) XS[j + 1][k * 32 + lane] = XS[j][k * 32 + lane] + GA[k * 32 + lane];
      __syncthreads();
    }
  }
  dense16(GA, TT, wsh + WSH_PW, wsh + WSH_PB, nullptr, lane, 0); // TT = t
  __syncthreads();

  // ---- backward ----
  const float* gg = gin + (size_t)n * XTILE;
  #pragma unroll
  for (int k = 0; k < 16; ++k) GT[k * 32 + lane] = gg[k * 32 + lane];
  // gsilu backward at TT -> GA
  G[lane] = 1.f / (1.f + __expf(-TT[lane]));
  {
    float s = 0.f;
    #pragma unroll
    for (int r = 0; r < 16; ++r) s += GT[r * 32 + lane] * TT[r * 32 + lane];
    CS[lane] = s;
  }
  #pragma unroll
  for (int r = 0; r < 16; ++r)
    GA[r * 32 + lane] = GT[r * 32 + lane] * G[lane]
                      + ((r == 0) ? CS[lane] * G[lane] * (1.f - G[lane]) : 0.f);
  __syncthreads();
  // g_y2 = GA @ Wp^T + GT
  dense16(GA, WK, wsh + WSH_PW, nullptr, nullptr, lane, 1);
  __syncthreads();
  #pragma unroll
  for (int r = 0; r < 16; ++r) WK[r * 32 + lane] += GT[r * 32 + lane];
  __syncthreads();
  for (int j = 2; j >= 0; --j) {
    const float* W0 = wsh + WSH_RESW + j * 2048;
    const float* W1 = W0 + 1024;
    if (j < 2) {                                   // g_y_j = current g_x_{j+1}
      #pragma unroll
      for (int r = 0; r < 16; ++r) WK[r * 32 + lane] = GT[r * 32 + lane];
      __syncthreads();
    }
    dense16(WK, GA, W1, nullptr, nullptr, lane, 1);   // g_u
    __syncthreads();
    {
      float gv = (vrow[wv][j][lane] > 0.f) ? 1.f : 0.f;  // grelu backward
      #pragma unroll
      for (int r = 0; r < 16; ++r) GA[r * 32 + lane] *= gv;
    }
    __syncthreads();
    dense16(GA, WK, W0, nullptr, nullptr, lane, 1);   // g_w
    __syncthreads();
    G[lane] = 1.f / (1.f + __expf(-XS[j][lane]));     // gsilu backward at XS[j]
    {
      float s = 0.f;
      #pragma unroll
      for (int r = 0; r < 16; ++r) s += WK[r * 32 + lane] * XS[j][r * 32 + lane];
      CS[lane] = s;
    }
    #pragma unroll
    for (int r = 0; r < 16; ++r)
      GT[r * 32 + lane] += WK[r * 32 + lane] * G[lane]
                         + ((r == 0) ? CS[lane] * G[lane] * (1.f - G[lane]) : 0.f);
    __syncthreads();
  }
  float* go = gout + (size_t)n * XTILE;
  #pragma unroll
  for (int k = 0; k < 16; ++k) go[k * 32 + lane] = GT[k * 32 + lane];
}

// ---------------------------------------------------------------------------
// Message backward (one wave per edge, lane = feature f). Grid is exact so
// no early-out; mp weights async-staged to LDS per block.
// For last=0 the upstream of xs is the embedding (no position dependence),
// so gxprev may be null and only geometry gradients contribute to forces.
// ---------------------------------------------------------------------------
__global__ __launch_bounds__(256) void k_msg_bwd(const float* __restrict__ xprev, const float* __restrict__ Yb,
                          const float* __restrict__ radb, const float* __restrict__ dispb,
                          const float* __restrict__ rb,
                          const int* __restrict__ srcv, const int* __restrict__ dstv,
                          const float* __restrict__ w1, const float* __restrict__ w2,
                          const float* __restrict__ gin, float* __restrict__ gxprev,
                          float* __restrict__ gpos, int last) {
  __shared__ float wsh[2048];
  int tid = threadIdx.x;
  for (int i = tid; i < 1024; i += 256) { cp1_async(&wsh[i], &w1[i]); cp1_async(&wsh[1024 + i], &w2[i]); }
  cp_wait_async();
  __syncthreads();
  const float* s1w = wsh;
  const float* s2w = wsh + 1024;
  int e = blockIdx.x * (blockDim.x >> 5) + (tid >> 5);   // exact: NEDGE = 8 * gridDim
  int lane = tid & 31;
  int sa = srcv[e], da = dstv[e];
  const float* rp = radb + e * 16;
  float r1[4], r2[4];
  #pragma unroll
  for (int l = 0; l < 4; ++l) {
    float s1 = 0.f, s2 = 0.f;
    #pragma unroll
    for (int b = 0; b < 16; ++b) {
      float rv = rp[b];
      s1 += rv * s1w[(b * 4 + l) * 32 + lane];
      s2 += rv * s2w[(b * 4 + l) * 32 + lane];
    }
    r1[l] = s1; r2[l] = s2;
  }
  const float* xs = xprev + (size_t)sa * XTILE;
  const float* Ye = Yb + e * 16;
  float xs0 = xs[lane];
  float gYl[16];
  float gR1[4] = {0.f,0.f,0.f,0.f}, gR2[4] = {0.f,0.f,0.f,0.f};
  if (last) {
    float g0 = gin[(size_t)da * XTILE + lane];
    #pragma unroll
    for (int r = 0; r < 16; ++r) {
      int l = degf(r);
      float yv = Ye[r];
      float xsr = xs[r * 32 + lane];
      gYl[r] = g0 * xsr * r1[l];
      gR1[l] += g0 * yv * xsr;
      float gxs = g0 * yv * r1[l];
      if (r == 0) { gxs += g0 * r2[0]; gR2[0] += g0 * xs0; }
      if (gxprev) atomicAdd(&gxprev[(size_t)sa * XTILE + r * 32 + lane], gxs);
    }
  } else {
    #pragma unroll
    for (int r = 0; r < 16; ++r) {
      int l = degf(r);
      float g = gin[(size_t)da * XTILE + r * 32 + lane];
      float yv = Ye[r];
      float xsr = xs[r * 32 + lane];
      gYl[r] = g * r1[l] * xs0;
      gR1[l] += g * yv * xs0;
      gR2[l] += g * xsr;
      if (gxprev) {
        atomicAdd(&gxprev[(size_t)sa * XTILE + r * 32 + lane], g * r2[l]);
        atomicAdd(&gxprev[(size_t)sa * XTILE + lane], g * yv * r1[l]);
      }
    }
  }
  float gY[16];
  #pragma unroll
  for (int r = 0; r < 16; ++r) gY[r] = wsum32(gYl[r]);
  float gb[16];
  #pragma unroll
  for (int b = 0; b < 16; ++b) {
    float s = 0.f;
    #pragma unroll
    for (int l = 0; l < 4; ++l)
      s += gR1[l] * s1w[(b * 4 + l) * 32 + lane] + gR2[l] * s2w[(b * 4 + l) * 32 + lane];
    gb[b] = wsum32(s);
  }
  if (lane == 0) {
    float dx = dispb[e * 3], dy = dispb[e * 3 + 1], dz = dispb[e * 3 + 2];
    float r = rb[e];
    float ir = 1.f / fmaxf(r, 1e-9f);
    float ux = dx * ir, uy = dy * ir, uz = dz * ir;
    float gux, guy, guz;
    sph16_jac(ux, uy, uz, gY, &gux, &guy, &guz);
    float rad[16], drad[16];
    radial16_d(r, rad, drad);
    float gr = 0.f;
    #pragma unroll
    for (int k = 0; k < 16; ++k) gr += gb[k] * drad[k];
    float dotp = gux * ux + guy * uy + guz * uz;
    float gdx = (gux - dotp * ux) * ir + gr * ux;
    float gdy = (guy - dotp * uy) * ir + gr * uy;
    float gdz = (guz - dotp * uz) * ir + gr * uz;
    atomicAdd(&gpos[sa * 3 + 0], gdx);  atomicAdd(&gpos[sa * 3 + 1], gdy);  atomicAdd(&gpos[sa * 3 + 2], gdz);
    atomicAdd(&gpos[da * 3 + 0], -gdx); atomicAdd(&gpos[da * 3 + 1], -gdy); atomicAdd(&gpos[da * 3 + 2], -gdz);
  }
}

// forces = -grad * atom_mask ; copy q
__global__ void k_finalize(const float* __restrict__ gpos, const float* __restrict__ am,
                           const float* __restrict__ qb,
                           float* __restrict__ out_forces, float* __restrict__ out_q) {
  int n = blockIdx.x * blockDim.x + threadIdx.x;
  if (n >= NATOM) return;
  float m = am[n];
  out_forces[n * 3 + 0] = -gpos[n * 3 + 0] * m;
  out_forces[n * 3 + 1] = -gpos[n * 3 + 1] * m;
  out_forces[n * 3 + 2] = -gpos[n * 3 + 2] * m;
  out_q[n] = qb[n];
}

// ---------------------------------------------------------------------------
// Launcher
// ---------------------------------------------------------------------------
extern "C" void kernel_launch(void* const* d_in, const int* in_sizes, int n_in,
                              void* d_out, int out_size, void* d_ws, size_t ws_size,
                              hipStream_t stream) {
  const float* pos   = (const float*)d_in[0];
  const float* bm    = (const float*)d_in[1];
  const float* am    = (const float*)d_in[2];
  const float* embed = (const float*)d_in[3];
  const float* mpw1  = (const float*)d_in[4];
  const float* mpw2  = (const float*)d_in[5];
  const float* resw  = (const float*)d_in[6];
  const float* resb  = (const float*)d_in[7];
  const float* postw = (const float*)d_in[8];
  const float* postb = (const float*)d_in[9];
  const float* qk    = (const float*)d_in[10];
  const float* cb    = (const float*)d_in[11];
  const float* ew1   = (const float*)d_in[12];
  const float* ew2   = (const float*)d_in[13];
  const float* eb    = (const float*)d_in[14];
  const int*   zi    = (const int*)d_in[15];
  const int*   dstv  = (const int*)d_in[16];
  const int*   srcv  = (const int*)d_in[17];
  const int*   seg   = (const int*)d_in[18];
  float* out = (float*)d_out;
  float* w = (float*)d_ws;

  // workspace layout (~53 MB of floats)
  size_t o = 0;
  float* dispb = w + o; o += (size_t)3 * NEDGE;
  float* rb    = w + o; o += NEDGE;
  float* Yb    = w + o; o += (size_t)16 * NEDGE;
  float* radb  = w + o; o += (size_t)16 * NEDGE;
  float* db    = w + o; o += NEDGE;
  float* Rb    = w + o; o += NEDGE;
  float* x_emb = w + o; o += XSZ;
  float* x_m0  = w + o; o += XSZ;
  float* x_a0  = w + o; o += XSZ;
  float* x_m1  = w + o; o += XSZ;
  float* x_a1  = w + o; o += XSZ;
  float* qb    = w + o; o += NATOM;
  float* ae    = w + o; o += NATOM;
  float* aele  = w + o; o += NATOM;
  float* gq    = w + o; o += NATOM;
  float* sbuf  = w + o; o += NATOM;
  float* gpos  = w + o; o += (size_t)3 * NATOM;
  float* g_a1  = w + o; o += XSZ;
  float* g_m1  = w + o; o += XSZ;
  float* g_a0  = w + o; o += XSZ;
  float* g_m0  = w + o; o += XSZ;
  (void)ws_size; (void)n_in; (void)in_sizes;

  float* out_energy = out;
  float* out_forces = out + NBATCH;
  float* out_q      = out + NBATCH + 3 * NATOM;
  float* out_bele   = out + NBATCH + 3 * NATOM + NATOM;

  const float* mpw1_0 = mpw1;                 const float* mpw2_0 = mpw2;
  const float* mpw1_1 = mpw1 + 16 * 4 * 32;   const float* mpw2_1 = mpw2 + 16 * 4 * 32;

  // zero accumulators + outputs
  k_fill0<<<(out_size + 255) / 256, 256, 0, stream>>>(out, out_size);
  k_fill0<<<(3 * NATOM + 255) / 256, 256, 0, stream>>>(gpos, 3 * NATOM);
  k_fill0<<<(NATOM + 255) / 256, 256, 0, stream>>>(gq, NATOM);
  k_fill0<<<(XSZ + 255) / 256, 256, 0, stream>>>(g_a0, XSZ);

  // forward
  k_edges<<<(NEDGE + 255) / 256, 256, 0, stream>>>(pos, bm, dstv, srcv, dispb, rb, Yb, radb, db, Rb);
  k_init_x<<<XSZ / 256, 256, 0, stream>>>(embed, zi, x_emb);
  k_msg<<<NATOM, 256, 0, stream>>>(x_emb, Yb, radb, srcv, mpw1_0, mpw2_0, x_m0, 0);
  k_mlp<<<NATOM / MLPW, 128, 0, stream>>>(x_m0, x_a0, resw, resb, postw, postb, 0);
  k_msg<<<NATOM, 256, 0, stream>>>(x_a0, Yb, radb, srcv, mpw1_1, mpw2_1, x_m1, 1);
  k_mlp<<<NATOM / MLPW, 128, 0, stream>>>(x_m1, x_a1, resw, resb, postw, postb, 1);
  k_readout<<<(NATOM + 255) / 256, 256, 0, stream>>>(x_a1, qk, cb, ew1, ew2, eb, zi, am, qb, ae, sbuf);
  k_elec<<<(NATOM + 255) / 256, 256, 0, stream>>>(qb, Rb, bm, am, ae, srcv, seg, aele, gq, out_energy, out_bele);

  // backward (forces)
  k_head_bwd<<<XSZ / 256, 256, 0, stream>>>(sbuf, gq, qk, ew1, ew2, am, g_a1);
  k_elec_bwd<<<(NEDGE + 255) / 256, 256, 0, stream>>>(qb, db, dispb, bm, am, srcv, dstv, gpos);
  k_mlp_bwd<<<NATOM / MLPBW, 64, 0, stream>>>(x_m1, g_a1, g_m1, resw, resb, postw, postb, 1);
  k_msg_bwd<<<NEDGE / 8, 256, 0, stream>>>(x_a0, Yb, radb, dispb, rb, srcv, dstv,
                                           mpw1_1, mpw2_1, g_m1, g_a0, gpos, 1);
  k_mlp_bwd<<<NATOM / MLPBW, 64, 0, stream>>>(x_m0, g_a0, g_m0, resw, resb, postw, postb, 0);
  k_msg_bwd<<<NEDGE / 8, 256, 0, stream>>>(x_emb, Yb, radb, dispb, rb, srcv, dstv,
                                           mpw1_0, mpw2_0, g_m0, nullptr, gpos, 0);
  k_finalize<<<(NATOM + 255) / 256, 256, 0, stream>>>(gpos, am, qb, out_forces, out_q);
}
